// PTTDecoder_61546881352368
// MI455X (gfx1250) — matrix operand order repaired and checked by reading, not compiled, not run on hardware
//
#include <hip/hip_runtime.h>
#include <hip/hip_bf16.h>

typedef _Float16 half16 __attribute__((ext_vector_type(16)));
typedef _Float16 half8  __attribute__((ext_vector_type(8)));
typedef float    float8 __attribute__((ext_vector_type(8)));

#define WMMA_F16(a, b, c) \
  __builtin_amdgcn_wmma_f32_16x16x32_f16(false, (a), false, (b), (short)0, (c), false, false)

// ---------------------------------------------------------------------------
// Fragment loaders (wave32); A and B share the contiguous k order below, not the ISA table order.
// A (16x32 f16): lane L -> m = L&15, h = L>>4; element e -> K = 16h + e
//   => one contiguous 16-element run at k0+16h
// B (32x16 f16, stored as W[N,K] row-major = B^T): lane L -> n = L&15, h = L>>4;
//   element e -> K = 16h + e  => one contiguous 16-half (32B, aligned) run.
// ---------------------------------------------------------------------------
static __device__ __forceinline__ half16 load_a_f32(const float* __restrict__ A, int lda,
                                                    int mBase, int k0) {
  const int lane = threadIdx.x & 31;
  const int m  = lane & 15;
  const int hh = lane >> 4;
  const float* row = A + (size_t)(mBase + m) * (size_t)lda + k0 + 16 * hh;
  half16 a;
#pragma unroll
  for (int e = 0; e < 8; ++e) {
    a[e]     = (_Float16)row[e];
    a[e + 8] = (_Float16)row[e + 8];
  }
  return a;
}

static __device__ __forceinline__ half16 load_a_f16(const _Float16* __restrict__ A, int lda,
                                                    int mBase, int k0) {
  const int lane = threadIdx.x & 31;
  const int m  = lane & 15;
  const int hh = lane >> 4;
  const _Float16* row = A + (size_t)(mBase + m) * (size_t)lda + k0 + 16 * hh;
  half8 lo = *(const half8*)row;          // 16B aligned
  half8 hi = *(const half8*)(row + 8);    // 16B aligned
  return __builtin_shufflevector(lo, hi, 0, 1, 2, 3, 4, 5, 6, 7,
                                 8, 9, 10, 11, 12, 13, 14, 15);
}

static __device__ __forceinline__ half16 load_b_f16(const _Float16* __restrict__ W, int ldw,
                                                    int nBase, int k0) {
  const int lane = threadIdx.x & 31;
  const int n  = lane & 15;
  const int hh = lane >> 4;
  const _Float16* p = W + (size_t)(nBase + n) * (size_t)ldw + k0 + 16 * hh;
  return *(const half16*)p;               // 32B aligned
}

// ---------------------------------------------------------------------------
// fp32 -> f16 weight conversion with zero padding (wte padded to 50304 rows)
// ---------------------------------------------------------------------------
__global__ void k_cvt_f16_pad(const float* __restrict__ src, _Float16* __restrict__ dst,
                              size_t nsrc, size_t ndst) {
  size_t stride = (size_t)gridDim.x * blockDim.x;
  for (size_t i = (size_t)blockIdx.x * blockDim.x + threadIdx.x; i < ndst; i += stride)
    dst[i] = (i < nsrc) ? (_Float16)src[i] : (_Float16)0.0f;
}

// h[bt, d] = wte[x[bt], d] + wpe[bt % T, d]
__global__ void k_embed(const int* __restrict__ x, const float* __restrict__ wte,
                        const float* __restrict__ wpe, float* __restrict__ h) {
  const int bt = blockIdx.x;
  const int d  = threadIdx.x;
  const int tok = x[bt];
  const int t = bt & 7; // T = 8
  h[(size_t)bt * 128 + d] = wte[(size_t)tok * 128 + d] + wpe[(size_t)t * 128 + d];
}

// C[M,N] = A[M,K] @ W[N,K]^T (+bias)(+ReLU).  Block: 16 rows x 128 cols, 8 waves.
__global__ __launch_bounds__(256)
void k_gemm(const float* __restrict__ A, const _Float16* __restrict__ W,
            const float* __restrict__ bias, float* __restrict__ C,
            int N, int K, int relu) {
  const int mBase = blockIdx.x * 16;
  const int wave  = threadIdx.x >> 5;
  const int nBase = blockIdx.y * 128 + wave * 16;
  float8 acc = {};
  for (int k0 = 0; k0 < K; k0 += 32) {
    half16 a = load_a_f32(A, K, mBase, k0);
    half16 b = load_b_f16(W, K, nBase, k0);
    acc = WMMA_F16(a, b, acc);
  }
  const int lane = threadIdx.x & 31;
  const int n  = nBase + (lane & 15);
  const int hh = lane >> 4;
  const float bv = bias ? bias[n] : 0.0f;
#pragma unroll
  for (int r = 0; r < 8; ++r) {
    float v = acc[r] + bv;
    if (relu) v = fmaxf(v, 0.0f);
    C[(size_t)(mBase + hh * 8 + r) * (size_t)N + n] = v;
  }
}

// h[m,:] += LN(A[m,:K] @ W[128,K]^T + bias) * g + b   (N == D == 128, in-place on h)
__global__ __launch_bounds__(256)
void k_gemm_ln_res(const float* __restrict__ A, const _Float16* __restrict__ W,
                   const float* __restrict__ bias, const float* __restrict__ g,
                   const float* __restrict__ bln, float* __restrict__ h, int K) {
  __shared__ float tile[16][132];   // padded to dodge bank conflicts
  __shared__ float red[2][16][16];
  __shared__ float stats[16][2];
  const int mBase = blockIdx.x * 16;
  const int wave  = threadIdx.x >> 5;
  const int nBase = wave * 16;
  float8 acc = {};
  for (int k0 = 0; k0 < K; k0 += 32) {
    half16 a = load_a_f32(A, K, mBase, k0);
    half16 b = load_b_f16(W, K, nBase, k0);
    acc = WMMA_F16(a, b, acc);
  }
  const int lane = threadIdx.x & 31;
  const int n  = nBase + (lane & 15);
  const int hh = lane >> 4;
  const float bv = bias[n];
#pragma unroll
  for (int r = 0; r < 8; ++r) tile[hh * 8 + r][n] = acc[r] + bv;
  __syncthreads();

  const int row = threadIdx.x >> 4;  // 16 rows x 16 segments
  const int seg = threadIdx.x & 15;
  float s = 0.0f, s2 = 0.0f;
#pragma unroll
  for (int e = 0; e < 8; ++e) {
    float v = tile[row][seg * 8 + e];
    s += v; s2 += v * v;
  }
  red[0][row][seg] = s;
  red[1][row][seg] = s2;
  __syncthreads();
  if (threadIdx.x < 16) {
    float ss = 0.0f, ss2 = 0.0f;
#pragma unroll
    for (int i = 0; i < 16; ++i) { ss += red[0][threadIdx.x][i]; ss2 += red[1][threadIdx.x][i]; }
    float mu  = ss * (1.0f / 128.0f);
    float var = ss2 * (1.0f / 128.0f) - mu * mu;
    stats[threadIdx.x][0] = mu;
    stats[threadIdx.x][1] = rsqrtf(var + 1e-5f);
  }
  __syncthreads();
  const float mu = stats[row][0];
  const float rs = stats[row][1];
#pragma unroll
  for (int e = 0; e < 8; ++e) {
    const int col = seg * 8 + e;
    const size_t idx = (size_t)(mBase + row) * 128 + col;
    h[idx] = h[idx] + (tile[row][col] - mu) * rs * g[col] + bln[col];
  }
}

// Final LayerNorm: one block (128 threads) per token row; emits f16 directly
// since the result feeds only the f16 logits WMMA.
__global__ __launch_bounds__(128)
void k_layernorm_f16(const float* __restrict__ in, _Float16* __restrict__ out,
                     const float* __restrict__ g, const float* __restrict__ b) {
  __shared__ float rs[128], rs2[128];
  const int row = blockIdx.x;
  const int d = threadIdx.x;
  const float v = in[(size_t)row * 128 + d];
  rs[d] = v; rs2[d] = v * v;
  __syncthreads();
  for (int off = 64; off > 0; off >>= 1) {
    if (d < off) { rs[d] += rs[d + off]; rs2[d] += rs2[d + off]; }
    __syncthreads();
  }
  const float mu  = rs[0] * (1.0f / 128.0f);
  const float var = rs2[0] * (1.0f / 128.0f) - mu * mu;
  const float r = rsqrtf(var + 1e-5f);
  out[(size_t)row * 128 + d] = (_Float16)((v - mu) * r * g[d] + b[d]);
}

// logits[2048, V] = hn16 @ wte16^T + fc_b.  Block: 64 rows x 128 cols, 8 waves;
// each wave reuses its B fragment across 4 A tiles (4x less L2 B-traffic).
// Inner loop is pure f16 b128-load -> wmma; nontemporal stores keep the 412MB
// logit stream from evicting wte16 from the 192MB L2.
__global__ __launch_bounds__(256)
void k_logits(const _Float16* __restrict__ A, const _Float16* __restrict__ Wt,
              const float* __restrict__ fcb, float* __restrict__ out, int V) {
  const int mBase = blockIdx.x * 64;
  const int wave  = threadIdx.x >> 5;
  const int nBase = blockIdx.y * 128 + wave * 16;
  float8 acc[4] = {};
#pragma unroll
  for (int k0 = 0; k0 < 128; k0 += 32) {
    half16 b = load_b_f16(Wt, 128, nBase, k0);
#pragma unroll
    for (int t = 0; t < 4; ++t) {
      half16 a = load_a_f16(A, 128, mBase + 16 * t, k0);
      acc[t] = WMMA_F16(a, b, acc[t]);
    }
  }
  const int lane = threadIdx.x & 31;
  const int n  = nBase + (lane & 15);
  const int hh = lane >> 4;
  if (n < V) {
    const float bv = fcb[n];
#pragma unroll
    for (int t = 0; t < 4; ++t)
#pragma unroll
      for (int r = 0; r < 8; ++r)
        __builtin_nontemporal_store(acc[t][r] + bv,
            &out[(size_t)(mBase + 16 * t + hh * 8 + r) * (size_t)V + n]);
  }
}

// ---------------------------------------------------------------------------
extern "C" void kernel_launch(void* const* d_in, const int* in_sizes, int n_in,
                              void* d_out, int out_size, void* d_ws, size_t ws_size,
                              hipStream_t stream) {
  const int L = 6, D = 128, DFF = 512, Vv = 50257, BT = 2048;
  const int NPAD = 50304; // 393 * 128

  const int*   x    = (const int*)  d_in[0];
  const float* wte  = (const float*)d_in[1];
  const float* wpe  = (const float*)d_in[2];
  const float* Wv   = (const float*)d_in[3];
  const float* Wo   = (const float*)d_in[4];
  const float* bo   = (const float*)d_in[5];
  const float* ln1g = (const float*)d_in[6];
  const float* ln1b = (const float*)d_in[7];
  const float* W1   = (const float*)d_in[8];
  const float* b1   = (const float*)d_in[9];
  const float* W2   = (const float*)d_in[10];
  const float* b2   = (const float*)d_in[11];
  const float* ln2g = (const float*)d_in[12];
  const float* ln2b = (const float*)d_in[13];
  const float* lnfg = (const float*)d_in[14];
  const float* lnfb = (const float*)d_in[15];
  const float* fcb  = (const float*)d_in[16];
  float* out = (float*)d_out;
  (void)in_sizes; (void)n_in; (void)out_size; (void)ws_size;

  // Workspace carve-out (~21.7 MB total, 256B aligned slices)
  char* ws = (char*)d_ws;
  size_t off = 0;
  auto carve = [&](size_t bytes) -> void* {
    void* p = ws + off;
    off += (bytes + 255) & ~(size_t)255;
    return p;
  };
  float*    h     = (float*)   carve((size_t)BT * D * 4);
  float*    vbuf  = (float*)   carve((size_t)BT * D * 4);
  float*    f1    = (float*)   carve((size_t)BT * DFF * 4);
  _Float16* hn16  = (_Float16*)carve((size_t)BT * D * 2);
  _Float16* Wv16  = (_Float16*)carve((size_t)L * D * D * 2);
  _Float16* Wo16  = (_Float16*)carve((size_t)L * D * D * 2);
  _Float16* W116  = (_Float16*)carve((size_t)L * DFF * D * 2);
  _Float16* W216  = (_Float16*)carve((size_t)L * D * DFF * 2);
  _Float16* wte16 = (_Float16*)carve((size_t)NPAD * D * 2);

  // One-time weight conversions (reused by 2048 token rows -> amortized)
  k_cvt_f16_pad<<<256,  256, 0, stream>>>(Wv,  Wv16,  (size_t)L*D*D,   (size_t)L*D*D);
  k_cvt_f16_pad<<<256,  256, 0, stream>>>(Wo,  Wo16,  (size_t)L*D*D,   (size_t)L*D*D);
  k_cvt_f16_pad<<<512,  256, 0, stream>>>(W1,  W116,  (size_t)L*DFF*D, (size_t)L*DFF*D);
  k_cvt_f16_pad<<<512,  256, 0, stream>>>(W2,  W216,  (size_t)L*D*DFF, (size_t)L*D*DFF);
  k_cvt_f16_pad<<<2048, 256, 0, stream>>>(wte, wte16, (size_t)Vv*D,    (size_t)NPAD*D);

  k_embed<<<BT, D, 0, stream>>>(x, wte, wpe, h);

  for (int l = 0; l < L; ++l) {
    // v = h @ Wv^T
    k_gemm<<<dim3(BT / 16, 1), 256, 0, stream>>>(
        h, Wv16 + (size_t)l * D * D, nullptr, vbuf, D, D, 0);
    // h += ln1(v @ Wo^T + bo)
    k_gemm_ln_res<<<BT / 16, 256, 0, stream>>>(
        vbuf, Wo16 + (size_t)l * D * D, bo + l * D, ln1g + l * D, ln1b + l * D, h, D);
    // f1 = relu(h @ W1^T + b1)
    k_gemm<<<dim3(BT / 16, DFF / 128), 256, 0, stream>>>(
        h, W116 + (size_t)l * DFF * D, b1 + l * DFF, f1, DFF, D, 1);
    // h += ln2(f1 @ W2^T + b2)
    k_gemm_ln_res<<<BT / 16, 256, 0, stream>>>(
        f1, W216 + (size_t)l * D * DFF, b2 + l * D, ln2g + l * D, ln2b + l * D, h, DFF);
  }

  k_layernorm_f16<<<BT, D, 0, stream>>>(h, hn16, lnfg, lnfb);
  k_logits<<<dim3(BT / 64, NPAD / 128), 256, 0, stream>>>(hn16, wte16, fcb, out, Vv);
}
